// NegativeSampling_86930138071091
// MI455X (gfx1250) — compile-verified, matrix operand order checked by reading, not run on hardware
//
#include <hip/hip_runtime.h>
#include <hip/hip_bf16.h>
#include <math.h>

typedef __attribute__((ext_vector_type(16))) _Float16 v16h;
typedef __attribute__((ext_vector_type(8)))  float    v8f;
typedef __attribute__((ext_vector_type(4)))  unsigned int v4u;
typedef __attribute__((ext_vector_type(8)))  int      v8i;
typedef __attribute__((ext_vector_type(4)))  int      v4i;

#define V_VOCAB 100000
#define N_POS   1024     // B*C*S
#define H_DIM   128
#define N_NEG   5
#define BATCH_B 32
#define ROWPAIRS 512     // N_POS/2 ; JAX threefry partner of (r,v) is (r+512,v)
#define CHUNK   4096     // logp elements staged per TDM tile (16 KB)
#define SUBIT   (CHUNK / 256)

#if defined(__has_builtin)
# if __has_builtin(__builtin_amdgcn_tensor_load_to_lds)
#  define HAVE_TDM 1
# endif
#endif
#ifndef HAVE_TDM
# define HAVE_TDM 0
#endif

// ---------------- threefry2x32, matching jax.random.key(1) ----------------
__device__ __forceinline__ uint32_t rotl32(uint32_t v, int d) {
  return (v << d) | (v >> (32 - d));
}

__device__ __forceinline__ void threefry2x32(uint32_t x0, uint32_t x1,
                                             uint32_t& o0, uint32_t& o1) {
  const uint32_t k0 = 0u, k1 = 1u;                 // PRNGKey(1) -> (0,1)
  const uint32_t k2 = k0 ^ k1 ^ 0x1BD11BDAu;
  x0 += k0; x1 += k1;
#define TF_ROUND(r) { x0 += x1; x1 = rotl32(x1, (r)); x1 ^= x0; }
  TF_ROUND(13) TF_ROUND(15) TF_ROUND(26) TF_ROUND(6)
  x0 += k1; x1 += k2 + 1u;
  TF_ROUND(17) TF_ROUND(29) TF_ROUND(16) TF_ROUND(24)
  x0 += k2; x1 += k0 + 2u;
  TF_ROUND(13) TF_ROUND(15) TF_ROUND(26) TF_ROUND(6)
  x0 += k0; x1 += k1 + 3u;
  TF_ROUND(17) TF_ROUND(29) TF_ROUND(16) TF_ROUND(24)
  x0 += k1; x1 += k2 + 4u;
  TF_ROUND(13) TF_ROUND(15) TF_ROUND(26) TF_ROUND(6)
  x0 += k2; x1 += k0 + 5u;
#undef TF_ROUND
  o0 = x0; o1 = x1;
}

__device__ __forceinline__ float bits_to_gumbel(uint32_t r) {
  // JAX: u = (bits>>9 | 0x3f800000) as float - 1  (in [0,1)), clamped to tiny
  float f = __uint_as_float((r >> 9) | 0x3f800000u) - 1.0f;
  const float tiny = 1.175494350822288e-38f;
  float u = fmaxf(f, tiny);
  return -logf(-logf(u));
}

__device__ __forceinline__ unsigned long long ullmax(unsigned long long a,
                                                     unsigned long long b) {
  return a > b ? a : b;
}

// ---------------- TDM helpers (Tensor Data Mover, gfx1250) ----------------
#if HAVE_TDM
// 1-D tile load: nelem_tile elements of 4B from gsrc into LDS at lds_off.
// tensor_dim0 = nelem_valid => reads past the valid range return zero (tail).
__device__ __forceinline__ void tdm_load_1d(const float* gsrc,
                                            uint32_t nelem_valid,
                                            uint32_t nelem_tile,
                                            uint32_t lds_off) {
  const uint64_t ga = (uint64_t)(uintptr_t)gsrc;
  v4u g0;
  g0.x = 1u;                                            // count=1, user descriptor
  g0.y = lds_off;                                       // lds_addr (bytes)
  g0.z = (uint32_t)ga;                                  // global_addr[31:0]
  g0.w = ((uint32_t)(ga >> 32) & 0x01FFFFFFu) | (2u << 30);  // addr[56:32] | type=2

  const uint32_t td0 = nelem_valid;                     // tensor_dim0 (OOB clamp)
  const uint32_t td1 = 1u;                              // tensor_dim1
  const uint32_t w0 = (2u << 16);                       // data_size=2 -> 4 bytes
  const uint32_t w1 = (td0 & 0xFFFFu) << 16;            // tensor_dim0[15:0]
  const uint32_t w2 = (td0 >> 16) | ((td1 & 0xFFFFu) << 16);
  const uint32_t w3 = (td1 >> 16) | ((nelem_tile & 0xFFFFu) << 16); // tile_dim0
  v8i g1 = (v8i){(int)w0, (int)w1, (int)w2, (int)w3, 0, 0, 0, 0};
  v4i z4 = (v4i){0, 0, 0, 0};
#if defined(__clang_major__) && (__clang_major__ >= 23)
  v8i z8 = (v8i){0, 0, 0, 0, 0, 0, 0, 0};
  __builtin_amdgcn_tensor_load_to_lds(g0, g1, z4, z4, z8, 0);
#else
  __builtin_amdgcn_tensor_load_to_lds(g0, g1, z4, z4, 0);
#endif
}

__device__ __forceinline__ void tdm_wait0() {
#if defined(__has_builtin)
# if __has_builtin(__builtin_amdgcn_s_wait_tensorcnt)
  __builtin_amdgcn_s_wait_tensorcnt(0);
# else
  asm volatile("s_wait_tensorcnt 0x0" ::: "memory");
# endif
#else
  asm volatile("s_wait_tensorcnt 0x0" ::: "memory");
#endif
}
#endif  // HAVE_TDM

// ------------- kernel A1: partial sums of wf^0.75 -------------
__global__ __launch_bounds__(256)
void ns_pow_partial(const float* __restrict__ wf, float* __restrict__ partials) {
  __shared__ float red[256];
  const int tid = threadIdx.x;
  float acc = 0.f;
  for (int v = blockIdx.x * 256 + tid; v < V_VOCAB; v += gridDim.x * 256)
    acc += powf(wf[v], 0.75f);
  red[tid] = acc; __syncthreads();
  for (int off = 128; off > 0; off >>= 1) {
    if (tid < off) red[tid] += red[tid + off];
    __syncthreads();
  }
  if (tid == 0) partials[blockIdx.x] = red[0];
}

// ------------- kernel A2: lnS = log(sum partials) -------------
__global__ __launch_bounds__(512)
void ns_finalize_lnS(const float* __restrict__ partials, float* __restrict__ lnS) {
  __shared__ float red[512];
  const int tid = threadIdx.x;
  red[tid] = partials[tid]; __syncthreads();
  for (int off = 256; off > 0; off >>= 1) {
    if (tid < off) red[tid] += red[tid + off];
    __syncthreads();
  }
  if (tid == 0) *lnS = logf(red[0]);
}

// ------------- kernel A3: logp[v] = 0.75*log(wf[v]) - lnS -------------
// Precomputed once so the hot kernel doesn't redo 51M logf's.
__global__ __launch_bounds__(256)
void ns_logp(const float* __restrict__ wf, const float* __restrict__ lnS_p,
             float* __restrict__ logp) {
  const int v = blockIdx.x * 256 + threadIdx.x;
  if (v < V_VOCAB) logp[v] = 0.75f * logf(wf[v]) - *lnS_p;
}

// ------------- kernel B: fused gumbel generation + per-row top-5 -------------
// One block per row-pair (r, r+512): one threefry call yields bits for both rows.
// logp is streamed through LDS in double-buffered 16KB tiles via the TDM.
__global__ __launch_bounds__(256)
void ns_gumbel_topk(const float* __restrict__ logp,
                    const int* __restrict__ target, int* __restrict__ negidx) {
  __shared__ float stage[2][CHUNK];              // 32 KB staging
  __shared__ unsigned long long cand[256 * 5];   // 10 KB merge candidates
  __shared__ unsigned long long red[256];

  const int tid = threadIdx.x;
  const int rp  = blockIdx.x;                    // 0..511
  const int t0  = target[rp];
  const int t1  = target[rp + ROWPAIRS];

  float s0[5], s1[5]; int i0[5], i1[5];
#pragma unroll
  for (int j = 0; j < 5; ++j) { s0[j] = s1[j] = -__builtin_inff(); i0[j] = i1[j] = 0; }

  const int nchunks = (V_VOCAB + CHUNK - 1) / CHUNK;   // 25

  // ---- stage chunk 0 ----
#if HAVE_TDM
  if (tid < 32)
    tdm_load_1d(logp, (uint32_t)V_VOCAB, CHUNK,
                (uint32_t)(uintptr_t)&stage[0][0]);
  if (tid < 32) tdm_wait0();
#else
  for (int i = tid; i < CHUNK; i += 256)
    stage[0][i] = (i < V_VOCAB) ? logp[i] : 0.f;
#endif
  __syncthreads();

  for (int c = 0; c < nchunks; ++c) {
    const int base = c * CHUNK;
    // ---- prefetch next chunk into the other buffer (async) ----
    if (c + 1 < nchunks) {
      const int nb = (c + 1) & 1;
#if HAVE_TDM
      if (tid < 32)
        tdm_load_1d(logp + (base + CHUNK),
                    (uint32_t)(V_VOCAB - (base + CHUNK)), CHUNK,
                    (uint32_t)(uintptr_t)&stage[nb][0]);
#else
      for (int i = tid; i < CHUNK; i += 256) {
        const int g = base + CHUNK + i;
        stage[nb][i] = (g < V_VOCAB) ? logp[g] : 0.f;
      }
#endif
    }

    // ---- consume current buffer ----
    const float* lbuf = stage[c & 1];
#pragma unroll 1
    for (int j = 0; j < SUBIT; ++j) {
      const int v = base + j * 256 + tid;
      if (v < V_VOCAB) {
        const float lp = lbuf[j * 256 + tid];
        uint32_t r0, r1;
        threefry2x32((uint32_t)(rp * V_VOCAB + v),
                     (uint32_t)((rp + ROWPAIRS) * V_VOCAB + v), r0, r1);
        const float sc0 = (v == t0) ? -__builtin_inff() : lp + bits_to_gumbel(r0);
        const float sc1 = (v == t1) ? -__builtin_inff() : lp + bits_to_gumbel(r1);
        // branchless sorted insert (descending); static indices -> stays in VGPRs
        {
          float cv = sc0; int ci = v;
#pragma unroll
          for (int q = 0; q < 5; ++q) {
            bool gt = cv > s0[q];
            float ts = s0[q]; int ti = i0[q];
            s0[q] = gt ? cv : ts;  i0[q] = gt ? ci : ti;
            cv    = gt ? ts : cv;  ci    = gt ? ti : ci;
          }
        }
        {
          float cv = sc1; int ci = v;
#pragma unroll
          for (int q = 0; q < 5; ++q) {
            bool gt = cv > s1[q];
            float ts = s1[q]; int ti = i1[q];
            s1[q] = gt ? cv : ts;  i1[q] = gt ? ci : ti;
            cv    = gt ? ts : cv;  ci    = gt ? ti : ci;
          }
        }
      }
    }

#if HAVE_TDM
    if (tid < 32) tdm_wait0();          // next chunk landed
#endif
    __syncthreads();                    // hand buffers over to all waves
  }

  // ----- merge 256 x top5 -> global top5 per row, via LDS max-extract -----
  for (int row = 0; row < 2; ++row) {
#pragma unroll
    for (int j = 0; j < 5; ++j) {
      float sv = row ? s1[j] : s0[j];
      int   iv = row ? i1[j] : i0[j];
      uint32_t b = __float_as_uint(sv);
      uint32_t key = (b & 0x80000000u) ? ~b : (b | 0x80000000u);  // order-preserving
      cand[tid * 5 + j] = ((unsigned long long)key << 32) | (uint32_t)iv;
    }
    __syncthreads();
    const int orow = rp + row * ROWPAIRS;
    for (int k = 0; k < N_NEG; ++k) {
      unsigned long long m = 0ull;
#pragma unroll
      for (int j = 0; j < 5; ++j) m = ullmax(m, cand[tid * 5 + j]);
      red[tid] = m; __syncthreads();
      for (int off = 128; off > 0; off >>= 1) {
        if (tid < off) red[tid] = ullmax(red[tid], red[tid + off]);
        __syncthreads();
      }
      const unsigned long long best = red[0];
#pragma unroll
      for (int j = 0; j < 5; ++j)
        if (cand[tid * 5 + j] == best) cand[tid * 5 + j] = 0ull;  // remove winner
      if (tid == 0) negidx[orow * N_NEG + k] = (int)(best & 0xffffffffu);
      __syncthreads();
    }
    __syncthreads();
  }
}

// ------------- kernel C: WMMA logits + BCE loss -------------
// One wave per 16-position tile. 6 slots (1 pos + 5 neg), each a 16x16 diag-GEMM:
// D = A(16x128 emb, f16) x B(128x16 gathered fc rows, f16), take diag.
__global__ __launch_bounds__(32)
void ns_loss(const float* __restrict__ emb, const int* __restrict__ target,
             const float* __restrict__ fc, const int* __restrict__ negidx,
             float* __restrict__ out) {
  const int lane  = threadIdx.x;        // 0..31
  const int p0    = blockIdx.x * 16;    // 64 tiles
  const int m     = lane & 15;          // A row / B column owned by this lane
  const int khalf = (lane < 16) ? 0 : 8;

  // A tile: 16-bit A layout (16x32 per chunk): lane m, VGPR j holds K pair
  // kp = (j<4 ? 2j : 16+2(j-4)) + khalf
  v16h a[4];
  const float* arow = emb + (size_t)(p0 + m) * H_DIM;
#pragma unroll
  for (int c = 0; c < 4; ++c) {
#pragma unroll
    for (int j = 0; j < 8; ++j) {
      const int kp = ((j < 4) ? 2 * j : 16 + 2 * (j - 4)) + khalf;
      const int k  = 32 * c + kp;
      a[c][2 * j]     = (_Float16)arow[k];
      a[c][2 * j + 1] = (_Float16)arow[k + 1];
    }
  }

  float lsum = 0.f;
#pragma unroll 1
  for (int s = 0; s < 6; ++s) {
    const int idx = (s == 0) ? target[p0 + m]
                             : negidx[(p0 + m) * N_NEG + (s - 1)];
    const float* brow = fc + (size_t)idx * H_DIM;
    v8f acc = {};
#pragma unroll
    for (int c = 0; c < 4; ++c) {
      v16h b;   // B 32x16 16-bit layout mirrors A with N=lane&15
#pragma unroll
      for (int j = 0; j < 8; ++j) {
        const int kp = ((j < 4) ? 2 * j : 16 + 2 * (j - 4)) + khalf;
        const int k  = 32 * c + kp;
        b[2 * j]     = (_Float16)brow[k];
        b[2 * j + 1] = (_Float16)brow[k + 1];
      }
      acc = __builtin_amdgcn_wmma_f32_16x16x32_f16(false, a[c], false, b,
                                                   (short)0, acc, false, false);
    }
    // Diagonal of 16x16 f32 C: M=0..7 -> lane m, vgpr m ; M=8..15 -> lane m+16, vgpr m-8
    int r = -1;
    if (lane < 8)        r = lane;
    else if (lane >= 24) r = lane - 24;
    if (r >= 0) {
      float d = 0.f;
#pragma unroll
      for (int q = 0; q < 8; ++q) if (q == r) d = acc[q];
      const float x  = (s == 0) ? -d : d;                       // softplus arg
      const float sp = fmaxf(x, 0.f) + log1pf(__expf(-fabsf(x)));
      lsum += (s == 0) ? sp : sp * (1.0f / BATCH_B);
    }
  }
  // wave32 reduction
  for (int off = 16; off > 0; off >>= 1)
    lsum += __shfl_down(lsum, off, 32);
  if (lane == 0) atomicAdd(out, lsum);
}

extern "C" void kernel_launch(void* const* d_in, const int* in_sizes, int n_in,
                              void* d_out, int out_size, void* d_ws, size_t ws_size,
                              hipStream_t stream) {
  const float* emb    = (const float*)d_in[0];   // (32,2,16,128)
  const int*   target = (const int*)d_in[1];     // (32,2,16)
  const float* fc     = (const float*)d_in[2];   // (100000,128)
  const float* wf     = (const float*)d_in[3];   // (100000,)
  float* out = (float*)d_out;

  float* partials = (float*)d_ws;                       // 512 floats
  float* lnS      = partials + 512;                     // 1 float (+pad)
  float* logp     = partials + 1024;                    // 100000 floats
  int*   negidx   = (int*)(logp + V_VOCAB);             // 1024*5 ints

  hipMemsetAsync(out, 0, sizeof(float), stream);
  ns_pow_partial <<<512, 256, 0, stream>>>(wf, partials);
  ns_finalize_lnS<<<1,   512, 0, stream>>>(partials, lnS);
  ns_logp        <<<(V_VOCAB + 255) / 256, 256, 0, stream>>>(wf, lnS, logp);
  ns_gumbel_topk <<<ROWPAIRS, 256, 0, stream>>>(logp, target, negidx);
  ns_loss        <<<N_POS / 16, 32, 0, stream>>>(emb, target, fc, negidx, out);
}
